// AKTModel_71365176591004
// MI455X (gfx1250) — compile-verified, matrix operand order checked by reading, not run on hardware
//
#include <hip/hip_runtime.h>
#include <hip/hip_fp16.h>
#include <math.h>

// ---------------------------------------------------------------------------
// AKT model for MI455X (gfx1250, wave32, WMMA 16x16x32 f16 -> f32, TDM)
// ---------------------------------------------------------------------------

typedef __attribute__((ext_vector_type(16))) _Float16 v16h;
typedef __attribute__((ext_vector_type(8)))  _Float16 v8h;
typedef __attribute__((ext_vector_type(8)))  float    v8f;

#define B_   16
#define Qn   512
#define Sn   511
#define Dm   128
#define Hn   8
#define KDn  16
#define FFn  256
#define Ln   4
#define NEGV (-1.0e9f)
#define SCALE_ 0.25f   // 1/sqrt(KD=16)

// --- Tensor Data Mover availability (device pass only) ---------------------
#if defined(__has_builtin)
#  if __has_builtin(__builtin_amdgcn_tensor_load_to_lds) && \
      __has_builtin(__builtin_amdgcn_s_wait_tensorcnt)
#    define AKT_TDM 1
#  else
#    define AKT_TDM 0
#  endif
#else
#  define AKT_TDM 0
#endif

__device__ __forceinline__ v8f wmma_f16(v16h a, v16h b, v8f c) {
    return __builtin_amdgcn_wmma_f32_16x16x32_f16(
        false, a, false, b, (short)0, c, false, false);
}

__device__ __forceinline__ float gelu_exact(float v) {
    return 0.5f * v * (1.0f + erff(v * 0.70710678118654752f));
}

#if AKT_TDM
typedef __attribute__((ext_vector_type(4))) unsigned int u32x4;
typedef __attribute__((ext_vector_type(8))) int i32x8;
typedef __attribute__((ext_vector_type(4))) int i32x4;

// DMA a 2D f16 tile (tile_h rows x tile_w elems, row stride = stride_elems)
// from global memory into LDS at byte offset lds_off. D# per ISA ch.8.
__device__ __forceinline__ void tdm_load_2d_f16(unsigned lds_off, const void* gptr,
                                                unsigned tile_w, unsigned tile_h,
                                                unsigned stride_elems)
{
    unsigned long long ga = (unsigned long long)(size_t)gptr;
    u32x4 g0;
    g0[0] = 1u;                                        // count=1 (valid user D#)
    g0[1] = lds_off;                                   // lds_addr (bytes)
    g0[2] = (unsigned)(ga & 0xffffffffu);              // global_addr[31:0]
    g0[3] = (unsigned)((ga >> 32) & 0x1ffffffu)        // global_addr[56:32]
          | (2u << 30);                                // type=2 ("image")
    i32x8 g1;
    g1[0] = (int)(1u << 16);                           // data_size=1 -> 2 bytes
    g1[1] = (int)(tile_w << 16);                       // tensor_dim0 lo
    g1[2] = (int)((tile_w >> 16) | (tile_h << 16));    // dim0 hi | tensor_dim1 lo
    g1[3] = (int)((tile_h >> 16) | (tile_w << 16));    // dim1 hi | tile_dim0
    g1[4] = (int)tile_h;                               // tile_dim1 (tile_dim2=0)
    g1[5] = (int)stride_elems;                         // tensor_dim0_stride lo
    g1[6] = 0;                                         // stride hi | dim1_stride
    g1[7] = 0;
    i32x4 z4 = {0, 0, 0, 0};
#if __clang_major__ >= 23
    i32x8 z8 = {0, 0, 0, 0, 0, 0, 0, 0};
    __builtin_amdgcn_tensor_load_to_lds(g0, g1, z4, z4, z8, 0);
#else
    __builtin_amdgcn_tensor_load_to_lds(g0, g1, z4, z4, 0);
#endif
}
#endif  // AKT_TDM

// ---------------------------------------------------------------------------
// GEMM: C[M,N] = act( A[M,K](f32) @ W[K,N] + bias (+R) ), W given as f16
// transposed panel Wt[N,K]. block = 128 thr (4 waves), tile 16(M) x 64(N).
// K templated (128/256) -> fully unrolled, one WMMA per wave per k-step.
// Weight panel arrives via TDM (tensor_load_to_lds) when available.
// ---------------------------------------------------------------------------
template<int K>
__global__ void __launch_bounds__(128)
k_gemm(const float* __restrict__ A, const __half* __restrict__ Wt,
       const float* __restrict__ bias, const float* __restrict__ R,
       float* __restrict__ C, int M, int N, int act)
{
    __shared__ _Float16 aT[16][32];     // A tile, row-major (m, k)
    __shared__ _Float16 bTt[64][32];    // W tile, (n, k) -- row-contiguous per col

    const int tid  = threadIdx.x;
    const int wave = tid >> 5;
    const int lane = tid & 31;
    const int m    = lane & 15;
    const int g    = lane >> 4;
    const int m0   = blockIdx.x * 16;
    const int n0   = blockIdx.y * 64;
    const int coln = wave * 16 + (lane & 15);
    const _Float16* WtH = (const _Float16*)Wt;

    v8f acc = {};

#pragma unroll
    for (int kt = 0; kt < K; kt += 32) {
#if AKT_TDM
        if (wave == 0) {
            unsigned lds_off = (unsigned)(size_t)&bTt[0][0];
            tdm_load_2d_f16(lds_off, WtH + (size_t)n0 * K + kt, 32, 64, (unsigned)K);
        }
#else
        for (int i = tid; i < 256; i += 128) {         // 64x32 halfs / 8
            int r = i >> 2, c8 = (i & 3) * 8;
            *(v8h*)&bTt[r][c8] = *(const v8h*)&WtH[(size_t)(n0 + r) * K + kt + c8];
        }
#endif
        if (kt + 32 < K)
            __builtin_prefetch(WtH + (size_t)n0 * K + kt + 32, 0, 1);

        // stage A 16x32 f32->f16: one float4 per thread
        {
            int r = tid >> 3, c = (tid & 7) * 4;
            const float4 av = *(const float4*)(A + (size_t)(m0 + r) * K + kt + c);
            __half2* dst = (__half2*)&aT[r][c];
            dst[0] = __floats2half2_rn(av.x, av.y);
            dst[1] = __floats2half2_rn(av.z, av.w);
        }
#if AKT_TDM
        if (wave == 0) __builtin_amdgcn_s_wait_tensorcnt(0);
#endif
        __syncthreads();

        // fragments: two contiguous 16B LDS reads each
        v16h af = __builtin_shufflevector(*(const v8h*)&aT[m][g * 8],
                                          *(const v8h*)&aT[m][16 + g * 8],
                                          0,1,2,3,4,5,6,7,8,9,10,11,12,13,14,15);
        v16h bf = __builtin_shufflevector(*(const v8h*)&bTt[coln][g * 8],
                                          *(const v8h*)&bTt[coln][16 + g * 8],
                                          0,1,2,3,4,5,6,7,8,9,10,11,12,13,14,15);
        acc = wmma_f16(af, bf, acc);
        __syncthreads();
    }

    const int col = n0 + coln;
#pragma unroll
    for (int r = 0; r < 8; ++r) {
        int row = m0 + r + 8 * g;
        float v = acc[r] + bias[col];
        if (R) v += R[(size_t)row * N + col];
        if (act == 1) v = gelu_exact(v);
        C[(size_t)row * N + col] = v;
    }
}

// ---------------------------------------------------------------------------
// Fused attention for one (b, h, 16-query tile). block = 128 thr (4 waves).
// scores = QK^T*SCALE + [causal - softplus(gamma)*dist]; softmax; P @ V.
// ---------------------------------------------------------------------------
__global__ void __launch_bounds__(128)
k_attn(const float* __restrict__ Qb, const float* __restrict__ Kb,
       const float* __restrict__ Vb, const float* __restrict__ gamma,
       float* __restrict__ Ob, int seq_q, int seq_kv)
{
    __shared__ float    sc[16][512];    // scores / probs (f32)
    __shared__ _Float16 vT[16][512];    // V^T head panel (kd, s) as f16
    __shared__ float    part[4][16][16];

    const int bh = blockIdx.x;
    const int b  = bh / Hn;
    const int h  = bh % Hn;
    const int q0 = blockIdx.y * 16;
    const int tid  = threadIdx.x;
    const int wave = tid >> 5;
    const int lane = tid & 31;
    const int m    = lane & 15;
    const int g    = lane >> 4;

    const float sp = log1pf(expf(gamma[h]));   // softplus(log_gamma)

    // --- stage V^T panel (512 x 16 -> vT[16][512]) cooperatively ---
    for (int s = tid; s < 512; s += 128) {
        if (s < seq_kv) {
            const float* vr = Vb + ((size_t)(b * seq_kv + s) * Dm) + h * KDn;
            float4 a0 = *(const float4*)(vr + 0);
            float4 a1 = *(const float4*)(vr + 4);
            float4 a2 = *(const float4*)(vr + 8);
            float4 a3 = *(const float4*)(vr + 12);
            float vv[16] = {a0.x,a0.y,a0.z,a0.w, a1.x,a1.y,a1.z,a1.w,
                            a2.x,a2.y,a2.z,a2.w, a3.x,a3.y,a3.z,a3.w};
#pragma unroll
            for (int kd = 0; kd < 16; ++kd) vT[kd][s] = (_Float16)vv[kd];
        } else {
#pragma unroll
            for (int kd = 0; kd < 16; ++kd) vT[kd][s] = (_Float16)0.0f;
        }
    }

    // --- Q fragment: 16 queries x 32 (kd 0..15 valid, rest zero-pad) ---
    v16h af = {};
    if ((q0 + m) < seq_q) {
        const float* qrow = Qb + ((size_t)(b * seq_q + q0 + m) * Dm) + h * KDn;
        float4 q1 = *(const float4*)(qrow + g * 8);
        float4 q2 = *(const float4*)(qrow + g * 8 + 4);
        af[0]=(_Float16)q1.x; af[1]=(_Float16)q1.y; af[2]=(_Float16)q1.z; af[3]=(_Float16)q1.w;
        af[4]=(_Float16)q2.x; af[5]=(_Float16)q2.y; af[6]=(_Float16)q2.z; af[7]=(_Float16)q2.w;
    }

    // --- scores: 32 key tiles of 16, round-robin over waves ---
    for (int st = wave; st < 32; st += 4) {
        v16h bf = {};
        const int scol = st * 16 + (lane & 15);
        if (scol < seq_kv) {
            const float* krow = Kb + ((size_t)(b * seq_kv + scol) * Dm) + h * KDn;
            float4 k1 = *(const float4*)(krow + g * 8);
            float4 k2 = *(const float4*)(krow + g * 8 + 4);
            bf[0]=(_Float16)k1.x; bf[1]=(_Float16)k1.y; bf[2]=(_Float16)k1.z; bf[3]=(_Float16)k1.w;
            bf[4]=(_Float16)k2.x; bf[5]=(_Float16)k2.y; bf[6]=(_Float16)k2.z; bf[7]=(_Float16)k2.w;
        }
        v8f c = {};
        c = wmma_f16(af, bf, c);           // wave-uniform: EXEC all ones

        const int cn = st * 16 + (lane & 15);
#pragma unroll
        for (int r = 0; r < 8; ++r) {
            int qi = q0 + r + 8 * g;
            float v = c[r] * SCALE_;
            v += (qi >= cn ? 0.0f : NEGV) - sp * fmaxf((float)(qi - cn), 0.0f);
            if (cn >= seq_kv) v = NEGV;
            sc[r + 8 * g][cn] = v;
        }
    }
    __syncthreads();

    // --- softmax: 4 rows per wave, 8 lanes per row ---
    {
        const int rr   = wave * 4 + (lane >> 3);
        const int lsub = lane & 7;
        float* row = sc[rr];
        float mx = -INFINITY;
        for (int s = lsub; s < 512; s += 8) mx = fmaxf(mx, row[s]);
#pragma unroll
        for (int o = 1; o < 8; o <<= 1) mx = fmaxf(mx, __shfl_xor(mx, o, 8));
        float sum = 0.0f;
        for (int s = lsub; s < 512; s += 8) { float e = expf(row[s] - mx); row[s] = e; sum += e; }
#pragma unroll
        for (int o = 1; o < 8; o <<= 1) sum += __shfl_xor(sum, o, 8);
        const float inv = 1.0f / sum;
        for (int s = lsub; s < 512; s += 8) row[s] *= inv;
    }
    __syncthreads();

    // --- output: O(16x16) = P(16x512) @ V(512x16); S split over waves by 32 ---
    v8f oacc = {};
    const int kd = lane & 15;
    for (int it = wave; it < 16; it += 4) {
        const int s0 = it * 32;
        float4 p0 = *(const float4*)&sc[m][s0 + g * 8];
        float4 p1 = *(const float4*)&sc[m][s0 + g * 8 + 4];
        float4 p2 = *(const float4*)&sc[m][s0 + 16 + g * 8];
        float4 p3 = *(const float4*)&sc[m][s0 + 16 + g * 8 + 4];
        v16h pa;
        pa[0]=(_Float16)p0.x;  pa[1]=(_Float16)p0.y;  pa[2]=(_Float16)p0.z;  pa[3]=(_Float16)p0.w;
        pa[4]=(_Float16)p1.x;  pa[5]=(_Float16)p1.y;  pa[6]=(_Float16)p1.z;  pa[7]=(_Float16)p1.w;
        pa[8]=(_Float16)p2.x;  pa[9]=(_Float16)p2.y;  pa[10]=(_Float16)p2.z; pa[11]=(_Float16)p2.w;
        pa[12]=(_Float16)p3.x; pa[13]=(_Float16)p3.y; pa[14]=(_Float16)p3.z; pa[15]=(_Float16)p3.w;
        v16h vb = __builtin_shufflevector(*(const v8h*)&vT[kd][s0 + g * 8],
                                          *(const v8h*)&vT[kd][s0 + 16 + g * 8],
                                          0,1,2,3,4,5,6,7,8,9,10,11,12,13,14,15);
        oacc = wmma_f16(pa, vb, oacc);
    }
#pragma unroll
    for (int r = 0; r < 8; ++r)
        part[wave][r + 8 * g][lane & 15] = oacc[r];
    __syncthreads();

    for (int i = tid; i < 256; i += 128) {
        int r = i >> 4, c = i & 15;
        float v = part[0][r][c] + part[1][r][c] + part[2][r][c] + part[3][r][c];
        int qi = q0 + r;
        if (qi < seq_q)
            Ob[((size_t)(b * seq_q + qi) * Dm) + h * KDn + c] = v;
    }
}

// ---------------------------------------------------------------------------
// LayerNorm over D=128: one wave per row.
// ---------------------------------------------------------------------------
__global__ void k_ln(const float* __restrict__ x, const float* __restrict__ s,
                     const float* __restrict__ bi, float* __restrict__ y)
{
    const int mrow = blockIdx.x;
    const int lane = threadIdx.x;
    const float* row = x + (size_t)mrow * Dm;
    float v[4], sum = 0.0f, sq = 0.0f;
#pragma unroll
    for (int i = 0; i < 4; ++i) {
        v[i] = row[lane + 32 * i];
        sum += v[i];
        sq  += v[i] * v[i];
    }
#pragma unroll
    for (int o = 16; o; o >>= 1) {
        sum += __shfl_xor(sum, o, 32);
        sq  += __shfl_xor(sq,  o, 32);
    }
    const float mean = sum * (1.0f / Dm);
    const float var  = sq * (1.0f / Dm) - mean * mean;
    const float inv  = rsqrtf(var + 1e-6f);
#pragma unroll
    for (int i = 0; i < 4; ++i) {
        int d = lane + 32 * i;
        y[(size_t)mrow * Dm + d] = (v[i] - mean) * inv * s[d] + bi[d];
    }
}

// weight convert+transpose: Wt[n,k] = (f16) W[k,n]
__global__ void k_wtr(const float* __restrict__ W, __half* __restrict__ Wt,
                      int K, int N)
{
    const int idx = blockIdx.x * blockDim.x + threadIdx.x;
    if (idx >= K * N) return;
    const int n = idx / K, k = idx % K;
    Wt[idx] = __float2half(W[(size_t)k * N + n]);
}

__global__ void k_embed_q(const int* __restrict__ questions,
                          const float* __restrict__ q_emb,
                          const float* __restrict__ rasch,
                          const float* __restrict__ pos_q,
                          float* __restrict__ q_repr)
{
    const int idx = blockIdx.x * blockDim.x + threadIdx.x;
    if (idx >= B_ * Qn * Dm) return;
    const int d = idx & (Dm - 1);
    const int t = idx / Dm;
    const int q = t % Qn;
    const int qid = questions[t];
    const float sg = 0.5f + 1.0f / (1.0f + expf(-rasch[qid]));
    q_repr[idx] = q_emb[(size_t)qid * Dm + d] * sg + pos_q[(size_t)q * Dm + d];
}

__global__ void k_embed_s(const int* __restrict__ iq, const int* __restrict__ ir,
                          const float* __restrict__ feat,
                          const float* __restrict__ q_emb, const float* __restrict__ diff_emb,
                          const float* __restrict__ rasch, const float* __restrict__ fp_w,
                          const float* __restrict__ fp_b, const float* __restrict__ resp_emb,
                          const float* __restrict__ ip_w, const float* __restrict__ ip_b,
                          const float* __restrict__ pos_s, const float* __restrict__ tp_w,
                          const float* __restrict__ tp_b, float* __restrict__ s_repr)
{
    __shared__ float comb[256];
    const int t = blockIdx.x;
    const int s = t % Sn;
    const int d = threadIdx.x;
    const int qid = iq[t];
    const int r   = ir[t];
    const float sg = 0.5f + 1.0f / (1.0f + expf(-rasch[qid]));

    comb[d] = q_emb[(size_t)qid * Dm + d] * sg;
    if (d < 32)
        comb[128 + d] = diff_emb[(size_t)qid * 32 + d] * (float)r;
    else if (d < 64)
        comb[128 + d] = resp_emb[(size_t)r * 32 + (d - 32)];
    else {
        int j = d - 64;
        float a = fp_b[j];
#pragma unroll
        for (int f = 0; f < 16; ++f) a += feat[(size_t)t * 16 + f] * fp_w[f * 64 + j];
        comb[192 + j] = a;
    }
    __syncthreads();

    float acc = ip_b[d];
    for (int k = 0; k < 256; ++k) acc += comb[k] * ip_w[(size_t)k * Dm + d];
    const float te = tanhf(feat[(size_t)t * 16] * tp_w[d] + tp_b[d]);
    s_repr[(size_t)t * Dm + d] = acc + pos_s[(size_t)s * Dm + d] + te;
}

__global__ void k_shift(const float* __restrict__ q_repr,
                        float* __restrict__ qq, float* __restrict__ qk)
{
    const int idx = blockIdx.x * blockDim.x + threadIdx.x;
    if (idx >= B_ * Sn * Dm) return;
    const int d = idx & (Dm - 1);
    const int t = idx / Dm;
    const int b = t / Sn;
    const int i = t % Sn;
    qq[idx] = q_repr[((size_t)(b * Qn + i + 1) * Dm) + d];
    qk[idx] = q_repr[((size_t)(b * Qn + i) * Dm) + d];
}

__global__ void k_head2(const float* __restrict__ X, const float* __restrict__ w,
                        const float* __restrict__ b, float* __restrict__ out)
{
    const int mrow = blockIdx.x;
    const int lane = threadIdx.x;
    float acc = 0.0f;
#pragma unroll
    for (int i = 0; i < 4; ++i) acc += X[(size_t)mrow * Dm + lane + 32 * i] * w[lane + 32 * i];
#pragma unroll
    for (int o = 16; o; o >>= 1) acc += __shfl_xor(acc, o, 32);
    if (lane == 0) out[mrow] = acc + b[0];
}

// ---------------------------------------------------------------------------
// host orchestration
// ---------------------------------------------------------------------------
extern "C" void kernel_launch(void* const* d_in, const int* in_sizes, int n_in,
                              void* d_out, int out_size, void* d_ws, size_t ws_size,
                              hipStream_t stream)
{
    (void)in_sizes; (void)n_in; (void)out_size; (void)ws_size;

    auto P = [&](int i) { return (const float*)d_in[i]; };
    const int*   questions = (const int*)d_in[0];
    const int*   inter_q   = (const int*)d_in[1];
    const int*   inter_r   = (const int*)d_in[2];
    const float* feat      = P(3);
    const float* q_emb = P(4);  const float* diff_emb = P(5);
    const float* rasch = P(6);  const float* fp_w = P(7);
    const float* fp_b  = P(8);  const float* resp_emb = P(9);
    const float* ip_w  = P(10); const float* ip_b = P(11);
    const float* pos_q = P(12); const float* pos_s = P(13);
    const float* tp_w  = P(14); const float* tp_b = P(15);
    const float* ow1   = P(16); const float* ob1 = P(17);
    const float* ow2   = P(18); const float* ob2 = P(19);

    // ---- workspace: fp32 region, then f16 (transposed) weight region ----
    float* ws = (float*)d_ws;
    size_t off = 0;
    auto alloc = [&](size_t n) { float* p = ws + off; off += n; return p; };
    float* q_repr = alloc((size_t)B_ * Qn * Dm);
    float* s_repr = alloc((size_t)B_ * Sn * Dm);
    float* qq     = alloc((size_t)B_ * Sn * Dm);
    float* qk     = alloc((size_t)B_ * Sn * Dm);
    float* qbuf   = alloc((size_t)B_ * Qn * Dm);
    float* kbuf   = alloc((size_t)B_ * Qn * Dm);
    float* vbuf   = alloc((size_t)B_ * Qn * Dm);
    float* obuf   = alloc((size_t)B_ * Qn * Dm);
    float* tmp    = alloc((size_t)B_ * Qn * Dm);
    float* xcur   = alloc((size_t)B_ * Qn * Dm);
    float* hbuf   = alloc((size_t)B_ * Qn * FFn);
    off = (off + 7) & ~(size_t)7;                 // 32B align for f16 area
    __half* hmem = (__half*)(ws + off);
    size_t hoff = 0;
    auto halloc = [&](size_t n) { __half* p = hmem + hoff; hoff += n; return p; };

    auto conv = [&](const float* W, __half* Wt, int K, int N) {
        int total = K * N;
        k_wtr<<<(total + 255) / 256, 256, 0, stream>>>(W, Wt, K, N);
    };

    // transposed f16 weights: [stack][layer][wq,wk,wv,wo,w1,w2] + ow1
    static const int bases[3] = {20, 37, 54};
    __half* WT[3][Ln][6];
    for (int s = 0; s < 3; ++s) {
        int base = bases[s];
        for (int l = 0; l < Ln; ++l) {
            WT[s][l][0] = halloc((size_t)Dm * Dm);
            WT[s][l][1] = halloc((size_t)Dm * Dm);
            WT[s][l][2] = halloc((size_t)Dm * Dm);
            WT[s][l][3] = halloc((size_t)Dm * Dm);
            WT[s][l][4] = halloc((size_t)Dm * FFn);
            WT[s][l][5] = halloc((size_t)FFn * Dm);
            conv(P(base + 0)  + (size_t)l * Dm * Dm,  WT[s][l][0], Dm,  Dm);
            conv(P(base + 2)  + (size_t)l * Dm * Dm,  WT[s][l][1], Dm,  Dm);
            conv(P(base + 4)  + (size_t)l * Dm * Dm,  WT[s][l][2], Dm,  Dm);
            conv(P(base + 6)  + (size_t)l * Dm * Dm,  WT[s][l][3], Dm,  Dm);
            conv(P(base + 9)  + (size_t)l * Dm * FFn, WT[s][l][4], Dm,  FFn);
            conv(P(base + 11) + (size_t)l * FFn * Dm, WT[s][l][5], FFn, Dm);
        }
    }
    __half* ow1T = halloc((size_t)Dm * Dm);
    conv(ow1, ow1T, Dm, Dm);

    auto gemm = [&](const float* A, const __half* Wt, const float* bias,
                    const float* R, float* C, int M, int N, int K, int act) {
        dim3 grid(M / 16, N / 64);
        if (K == 128) k_gemm<128><<<grid, 128, 0, stream>>>(A, Wt, bias, R, C, M, N, act);
        else          k_gemm<256><<<grid, 128, 0, stream>>>(A, Wt, bias, R, C, M, N, act);
    };

    // one transformer block; writes result back into xq
    auto run_block = [&](float* xq, const float* xk, const float* xv,
                         int sidx, int l, int seq_q, int seq_kv) {
        const int base = bases[sidx];
        const int Mq  = B_ * seq_q;
        const int Mkv = B_ * seq_kv;
        const float* bq = P(base + 1)  + (size_t)l * Dm;
        const float* bk = P(base + 3)  + (size_t)l * Dm;
        const float* bv = P(base + 5)  + (size_t)l * Dm;
        const float* bo = P(base + 7)  + (size_t)l * Dm;
        const float* gm = P(base + 8)  + (size_t)l * Hn;
        const float* b1 = P(base + 10) + (size_t)l * FFn;
        const float* b2 = P(base + 12) + (size_t)l * Dm;
        const float* l1s = P(base + 13) + (size_t)l * Dm;
        const float* l1b = P(base + 14) + (size_t)l * Dm;
        const float* l2s = P(base + 15) + (size_t)l * Dm;
        const float* l2b = P(base + 16) + (size_t)l * Dm;

        gemm(xq, WT[sidx][l][0], bq, nullptr, qbuf, Mq,  Dm, Dm, 0);
        gemm(xk, WT[sidx][l][1], bk, nullptr, kbuf, Mkv, Dm, Dm, 0);
        gemm(xv, WT[sidx][l][2], bv, nullptr, vbuf, Mkv, Dm, Dm, 0);
        k_attn<<<dim3(B_ * Hn, (seq_q + 15) / 16), 128, 0, stream>>>(
            qbuf, kbuf, vbuf, gm, obuf, seq_q, seq_kv);
        gemm(obuf, WT[sidx][l][3], bo, xq, tmp, Mq, Dm, Dm, 0);      // + residual
        k_ln<<<Mq, 32, 0, stream>>>(tmp, l1s, l1b, xcur);
        gemm(xcur, WT[sidx][l][4], b1, nullptr, hbuf, Mq, FFn, Dm, 1);  // GELU
        gemm(hbuf, WT[sidx][l][5], b2, xcur, tmp, Mq, Dm, FFn, 0);   // + residual
        k_ln<<<Mq, 32, 0, stream>>>(tmp, l2s, l2b, xq);              // write back
    };

    // embeddings / input projection
    {
        int total = B_ * Qn * Dm;
        k_embed_q<<<(total + 255) / 256, 256, 0, stream>>>(questions, q_emb, rasch,
                                                           pos_q, q_repr);
        k_embed_s<<<B_ * Sn, 128, 0, stream>>>(inter_q, inter_r, feat, q_emb, diff_emb,
                                               rasch, fp_w, fp_b, resp_emb, ip_w, ip_b,
                                               pos_s, tp_w, tp_b, s_repr);
    }

    // self-attention stacks
    for (int l = 0; l < Ln; ++l) run_block(q_repr, q_repr, q_repr, 0, l, Qn, Qn);
    for (int l = 0; l < Ln; ++l) run_block(s_repr, s_repr, s_repr, 1, l, Sn, Sn);

    // knowledge-retriever stack
    {
        int total = B_ * Sn * Dm;
        k_shift<<<(total + 255) / 256, 256, 0, stream>>>(q_repr, qq, qk);
    }
    for (int l = 0; l < Ln; ++l) run_block(qq, qk, s_repr, 2, l, Sn, Sn);

    // prediction head
    gemm(qq, ow1T, ob1, nullptr, hbuf, B_ * Sn, Dm, Dm, 1);          // GELU
    k_head2<<<B_ * Sn, 32, 0, stream>>>(hbuf, ow2, ob2, (float*)d_out);
}
